// ScaledDotProduct_18597208392077
// MI455X (gfx1250) — compile-verified, matrix operand order checked by reading, not run on hardware
//
#include <hip/hip_runtime.h>

typedef __bf16 bf16;
typedef __attribute__((ext_vector_type(16))) bf16  v16bf;
typedef __attribute__((ext_vector_type(8)))  float v8f;

#define BM 128
#define BN 128
#define BKK 32
#define LDT 40   // padded LDS row stride in bf16 elements -> 80B rows, 16B-aligned halves

// ---- CDNA5 async global->LDS path (ASYNCcnt), guarded so compile never regresses
#if defined(__has_builtin)
#if __has_builtin(__builtin_amdgcn_global_load_async_to_lds_b128)
#define HAS_ASYNC_LDS 1
#endif
#endif
#ifndef HAS_ASYNC_LDS
#define HAS_ASYNC_LDS 0
#endif

#if HAS_ASYNC_LDS
typedef int v4i __attribute__((vector_size(16)));
typedef __attribute__((address_space(1))) v4i* gv4i_p;   // global int4*
typedef __attribute__((address_space(3))) v4i* lv4i_p;   // LDS int4*
__device__ __forceinline__ void async_b128(const void* g, void* l) {
    __builtin_amdgcn_global_load_async_to_lds_b128((gv4i_p)g, (lv4i_p)l, 0, 0);
}
__device__ __forceinline__ void wait_async0() {
#if __has_builtin(__builtin_amdgcn_s_wait_asynccnt)
    __builtin_amdgcn_s_wait_asynccnt(0);
#else
    asm volatile("s_wait_asynccnt 0x0" ::: "memory");
#endif
}
#else
__device__ __forceinline__ void wait_async0() {}
#endif

union FragU { uint4 u[2]; v16bf v; };

// Load a 16-element bf16 fragment (32B) from LDS via two 16B-aligned b128 reads.
__device__ __forceinline__ v16bf ldsFrag(const bf16* p) {
    FragU f;
    const uint4* q = (const uint4*)p;
    f.u[0] = q[0];
    f.u[1] = q[1];
    return f.v;
}

// ---------------------------------------------------------------------------
// absmax reduction: one launch per weight matrix
// ---------------------------------------------------------------------------
__global__ void __launch_bounds__(256) init_scales_kernel(unsigned* scales) {
    if (threadIdx.x < 3) scales[threadIdx.x] = 0u;
}

__global__ void __launch_bounds__(256)
absmax_kernel(const float* __restrict__ W, unsigned* __restrict__ out, int n) {
    int base = blockIdx.x * 1024 + threadIdx.x;
    float m = 0.0f;
    #pragma unroll
    for (int i = 0; i < 4; ++i) {
        int id = base + i * 256;
        if (id < n) m = fmaxf(m, fabsf(W[id]));
    }
    #pragma unroll
    for (int o = 16; o > 0; o >>= 1) m = fmaxf(m, __shfl_xor(m, o, 32));
    __shared__ float red[8];
    if ((threadIdx.x & 31) == 0) red[threadIdx.x >> 5] = m;
    __syncthreads();
    if (threadIdx.x == 0) {
        float bm = red[0];
        #pragma unroll
        for (int i = 1; i < 8; ++i) bm = fmaxf(bm, red[i]);
        atomicMax(out, __float_as_uint(bm));  // abs values: uint order == float order
    }
}

// ---------------------------------------------------------------------------
// Fake-quantize W (round(W/s)*s, s = absmax/128) and store transposed bf16:
// Wt[n*D + k] = quant(W[k*D + n])
// ---------------------------------------------------------------------------
__global__ void __launch_bounds__(256)
quantize_t_kernel(const float* __restrict__ W, const unsigned* __restrict__ amaxbits,
                  bf16* __restrict__ Wt, int D) {
    float amax = __uint_as_float(*amaxbits);
    float s    = amax * (1.0f / 128.0f);        // 2*amax/256
    float inv  = (s > 0.0f) ? 1.0f / s : 0.0f;
    int idx = blockIdx.x * 256 + threadIdx.x;   // D*D threads total
    int k = idx >> 10;      // row of W (D == 1024)
    int n = idx & 1023;     // col of W
    float q = rintf(W[idx] * inv) * s;
    Wt[(size_t)n * D + k] = (bf16)q;
}

// ---------------------------------------------------------------------------
// GEMM: C = A(f32)[M,K] * Bt(bf16)[N,K]^T + bias, output bf16.
// A tile register-staged (f32->bf16 convert); B tile async-copied to LDS.
// Double-buffered LDS, one barrier per K-step.
// transOut: store Out[b][n][s] (vp^T layout) instead of Out[m][n]; m = b*2048+s.
// ---------------------------------------------------------------------------
__global__ void __launch_bounds__(256)
gemm_xw_kernel(const float* __restrict__ A, const bf16* __restrict__ Bt,
               const float* __restrict__ bias, bf16* __restrict__ Out,
               int M, int N, int K, int transOut) {
    __shared__ alignas(16) bf16 As[2][BM * LDT];
    __shared__ alignas(16) bf16 Bs[2][BN * LDT];

    const int tid  = threadIdx.x;
    const int lane = tid & 31;
    const int w    = tid >> 5;
    const int wm   = w >> 1;      // 0..3
    const int wn   = w & 1;       // 0..1
    const int m0g  = blockIdx.y * BM;
    const int n0g  = blockIdx.x * BN;

    const int sr = tid >> 1;            // 0..127
    const int sc = (tid & 1) * 16;      // 0 or 16
    const int ldsOff = sr * LDT + sc;

    v8f acc[2][4];
    #pragma unroll
    for (int i = 0; i < 2; ++i)
        #pragma unroll
        for (int j = 0; j < 4; ++j)
            #pragma unroll
            for (int e = 0; e < 8; ++e) acc[i][j][e] = 0.0f;

    const float* aG = A  + (size_t)(m0g + sr) * K + sc;
    const bf16*  bG = Bt + (size_t)(n0g + sr) * K + sc;

    const int lr = lane & 15;           // M/N index within tile
    const int kh = (lane >> 4) * 16;    // K half select

    float4 ra0, ra1, ra2, ra3;
    uint4  rb0, rb1;

    // ---- prologue: stage tile kk=0 into buffer 0
    ra0 = *(const float4*)(aG + 0);
    ra1 = *(const float4*)(aG + 4);
    ra2 = *(const float4*)(aG + 8);
    ra3 = *(const float4*)(aG + 12);
#if HAS_ASYNC_LDS
    async_b128(bG + 0, &Bs[0][ldsOff]);
    async_b128(bG + 8, &Bs[0][ldsOff + 8]);
#else
    rb0 = *(const uint4*)(bG + 0);
    rb1 = *(const uint4*)(bG + 8);
#endif
    {
        FragU fa;
        bf16* ap = (bf16*)&fa;
        ap[0]=(bf16)ra0.x; ap[1]=(bf16)ra0.y; ap[2]=(bf16)ra0.z; ap[3]=(bf16)ra0.w;
        ap[4]=(bf16)ra1.x; ap[5]=(bf16)ra1.y; ap[6]=(bf16)ra1.z; ap[7]=(bf16)ra1.w;
        ap[8]=(bf16)ra2.x; ap[9]=(bf16)ra2.y; ap[10]=(bf16)ra2.z; ap[11]=(bf16)ra2.w;
        ap[12]=(bf16)ra3.x; ap[13]=(bf16)ra3.y; ap[14]=(bf16)ra3.z; ap[15]=(bf16)ra3.w;
        uint4* d = (uint4*)&As[0][ldsOff];
        d[0] = fa.u[0]; d[1] = fa.u[1];
#if !HAS_ASYNC_LDS
        uint4* db = (uint4*)&Bs[0][ldsOff];
        db[0] = rb0; db[1] = rb1;
#endif
    }
    wait_async0();
    __syncthreads();

    int buf = 0;
    for (int kk = 0; kk < K; kk += BKK) {
        const bool hasNext = (kk + BKK) < K;
        // ---- issue next tile's loads (async B goes straight to LDS buf^1)
        if (hasNext) {
            const int kn = kk + BKK;
            ra0 = *(const float4*)(aG + kn + 0);
            ra1 = *(const float4*)(aG + kn + 4);
            ra2 = *(const float4*)(aG + kn + 8);
            ra3 = *(const float4*)(aG + kn + 12);
#if HAS_ASYNC_LDS
            async_b128(bG + kn + 0, &Bs[buf ^ 1][ldsOff]);
            async_b128(bG + kn + 8, &Bs[buf ^ 1][ldsOff + 8]);
#else
            rb0 = *(const uint4*)(bG + kn + 0);
            rb1 = *(const uint4*)(bG + kn + 8);
            __builtin_prefetch(aG + kn + BKK, 0, 1);
#endif
        }

        // ---- compute on current buffer
        v16bf af0 = ldsFrag(&As[buf][(wm * 32 + 0  + lr) * LDT + kh]);
        v16bf af1 = ldsFrag(&As[buf][(wm * 32 + 16 + lr) * LDT + kh]);
        #pragma unroll
        for (int j = 0; j < 4; ++j) {
            v16bf bfr = ldsFrag(&Bs[buf][(wn * 64 + j * 16 + lr) * LDT + kh]);
            acc[0][j] = __builtin_amdgcn_wmma_f32_16x16x32_bf16(
                false, af0, false, bfr, (short)0, acc[0][j], false, false);
            acc[1][j] = __builtin_amdgcn_wmma_f32_16x16x32_bf16(
                false, af1, false, bfr, (short)0, acc[1][j], false, false);
        }

        // ---- commit register-staged data into buf^1
        if (hasNext) {
            FragU fa;
            bf16* ap = (bf16*)&fa;
            ap[0]=(bf16)ra0.x; ap[1]=(bf16)ra0.y; ap[2]=(bf16)ra0.z; ap[3]=(bf16)ra0.w;
            ap[4]=(bf16)ra1.x; ap[5]=(bf16)ra1.y; ap[6]=(bf16)ra1.z; ap[7]=(bf16)ra1.w;
            ap[8]=(bf16)ra2.x; ap[9]=(bf16)ra2.y; ap[10]=(bf16)ra2.z; ap[11]=(bf16)ra2.w;
            ap[12]=(bf16)ra3.x; ap[13]=(bf16)ra3.y; ap[14]=(bf16)ra3.z; ap[15]=(bf16)ra3.w;
            uint4* d = (uint4*)&As[buf ^ 1][ldsOff];
            d[0] = fa.u[0]; d[1] = fa.u[1];
#if !HAS_ASYNC_LDS
            uint4* db = (uint4*)&Bs[buf ^ 1][ldsOff];
            db[0] = rb0; db[1] = rb1;
#endif
        }
        wait_async0();
        __syncthreads();
        buf ^= 1;
    }

    // Epilogue: C layout — VGPR r: m = r + 8*(lane>=16), n = lane&15
    const int mrow0 = m0g + wm * 32 + ((lane >> 4) << 3);
    const int ncol0 = n0g + wn * 64 + lr;
    #pragma unroll
    for (int i = 0; i < 2; ++i) {
        #pragma unroll
        for (int j = 0; j < 4; ++j) {
            int n = ncol0 + j * 16;
            float bv = bias[n];
            #pragma unroll
            for (int r = 0; r < 8; ++r) {
                int m = mrow0 + i * 16 + r;
                float vsum = acc[i][j][r] + bv;
                if (!transOut) {
                    Out[(size_t)m * N + n] = (bf16)vsum;
                } else {
                    int bb = m >> 11, ss = m & 2047;   // S == 2048
                    Out[((size_t)bb * N + n) * 2048 + ss] = (bf16)vsum;
                }
            }
        }
    }
}

// ---------------------------------------------------------------------------
// GEMM: C(f32)[M,N] = alpha * A(bf16)[M,K] * Bt(bf16)[N,K]^T
// Both tiles async-copied global->LDS; double-buffered, one barrier per step.
// ---------------------------------------------------------------------------
__global__ void __launch_bounds__(256)
gemm_bf16_kernel(const bf16* __restrict__ A, const bf16* __restrict__ Bt,
                 float* __restrict__ C, int M, int N, int K, float alpha) {
    __shared__ alignas(16) bf16 As[2][BM * LDT];
    __shared__ alignas(16) bf16 Bs[2][BN * LDT];

    const int tid  = threadIdx.x;
    const int lane = tid & 31;
    const int w    = tid >> 5;
    const int wm   = w >> 1;
    const int wn   = w & 1;
    const int m0g  = blockIdx.y * BM;
    const int n0g  = blockIdx.x * BN;
    const int sr   = tid >> 1;
    const int sc   = (tid & 1) * 16;
    const int ldsOff = sr * LDT + sc;

    v8f acc[2][4];
    #pragma unroll
    for (int i = 0; i < 2; ++i)
        #pragma unroll
        for (int j = 0; j < 4; ++j)
            #pragma unroll
            for (int e = 0; e < 8; ++e) acc[i][j][e] = 0.0f;

    const bf16* aG = A  + (size_t)(m0g + sr) * K + sc;
    const bf16* bG = Bt + (size_t)(n0g + sr) * K + sc;

    const int lr = lane & 15;
    const int kh = (lane >> 4) * 16;

    uint4 ra0, ra1, rb0, rb1;

    // ---- prologue: stage tile kk=0 into buffer 0
#if HAS_ASYNC_LDS
    async_b128(aG + 0, &As[0][ldsOff]);
    async_b128(aG + 8, &As[0][ldsOff + 8]);
    async_b128(bG + 0, &Bs[0][ldsOff]);
    async_b128(bG + 8, &Bs[0][ldsOff + 8]);
#else
    ra0 = *(const uint4*)(aG + 0);
    ra1 = *(const uint4*)(aG + 8);
    rb0 = *(const uint4*)(bG + 0);
    rb1 = *(const uint4*)(bG + 8);
    { uint4* d = (uint4*)&As[0][ldsOff]; d[0] = ra0; d[1] = ra1; }
    { uint4* d = (uint4*)&Bs[0][ldsOff]; d[0] = rb0; d[1] = rb1; }
#endif
    wait_async0();
    __syncthreads();

    int buf = 0;
    for (int kk = 0; kk < K; kk += BKK) {
        const bool hasNext = (kk + BKK) < K;
        if (hasNext) {
            const int kn = kk + BKK;
#if HAS_ASYNC_LDS
            async_b128(aG + kn + 0, &As[buf ^ 1][ldsOff]);
            async_b128(aG + kn + 8, &As[buf ^ 1][ldsOff + 8]);
            async_b128(bG + kn + 0, &Bs[buf ^ 1][ldsOff]);
            async_b128(bG + kn + 8, &Bs[buf ^ 1][ldsOff + 8]);
#else
            ra0 = *(const uint4*)(aG + kn + 0);
            ra1 = *(const uint4*)(aG + kn + 8);
            rb0 = *(const uint4*)(bG + kn + 0);
            rb1 = *(const uint4*)(bG + kn + 8);
#endif
        }

        v16bf af0 = ldsFrag(&As[buf][(wm * 32 + 0  + lr) * LDT + kh]);
        v16bf af1 = ldsFrag(&As[buf][(wm * 32 + 16 + lr) * LDT + kh]);
        #pragma unroll
        for (int j = 0; j < 4; ++j) {
            v16bf bfr = ldsFrag(&Bs[buf][(wn * 64 + j * 16 + lr) * LDT + kh]);
            acc[0][j] = __builtin_amdgcn_wmma_f32_16x16x32_bf16(
                false, af0, false, bfr, (short)0, acc[0][j], false, false);
            acc[1][j] = __builtin_amdgcn_wmma_f32_16x16x32_bf16(
                false, af1, false, bfr, (short)0, acc[1][j], false, false);
        }

#if !HAS_ASYNC_LDS
        if (hasNext) {
            { uint4* d = (uint4*)&As[buf ^ 1][ldsOff]; d[0] = ra0; d[1] = ra1; }
            { uint4* d = (uint4*)&Bs[buf ^ 1][ldsOff]; d[0] = rb0; d[1] = rb1; }
        }
#endif
        wait_async0();
        __syncthreads();
        buf ^= 1;
    }

    const int mrow0 = m0g + wm * 32 + ((lane >> 4) << 3);
    const int ncol0 = n0g + wn * 64 + lr;
    #pragma unroll
    for (int i = 0; i < 2; ++i)
        #pragma unroll
        for (int j = 0; j < 4; ++j)
            #pragma unroll
            for (int r = 0; r < 8; ++r) {
                int m = mrow0 + i * 16 + r;
                int n = ncol0 + j * 16;
                C[(size_t)m * N + n] = acc[i][j][r] * alpha;
            }
}

// ---------------------------------------------------------------------------
// Row softmax over 2048 f32 logits -> bf16 attn. One 256-thread block per row.
// ---------------------------------------------------------------------------
__global__ void __launch_bounds__(256)
softmax_kernel(const float* __restrict__ L, bf16* __restrict__ Outp, int cols) {
    const int row = blockIdx.x;
    const float* lp = L + (size_t)row * cols;
    const int tid = threadIdx.x;

    float v[8];
    float mx = -3.0e38f;
    #pragma unroll
    for (int i = 0; i < 8; ++i) {
        v[i] = lp[tid + i * 256];
        mx = fmaxf(mx, v[i]);
    }
    #pragma unroll
    for (int o = 16; o > 0; o >>= 1) mx = fmaxf(mx, __shfl_xor(mx, o, 32));
    __shared__ float redm[8];
    if ((tid & 31) == 0) redm[tid >> 5] = mx;
    __syncthreads();
    float bmx = redm[0];
    #pragma unroll
    for (int i = 1; i < 8; ++i) bmx = fmaxf(bmx, redm[i]);

    float s = 0.0f;
    #pragma unroll
    for (int i = 0; i < 8; ++i) {
        v[i] = __expf(v[i] - bmx);
        s += v[i];
    }
    #pragma unroll
    for (int o = 16; o > 0; o >>= 1) s += __shfl_xor(s, o, 32);
    __shared__ float reds[8];
    if ((tid & 31) == 0) reds[tid >> 5] = s;
    __syncthreads();
    float bs = 0.0f;
    #pragma unroll
    for (int i = 0; i < 8; ++i) bs += reds[i];
    float inv = 1.0f / bs;

    bf16* op = Outp + (size_t)row * cols;
    #pragma unroll
    for (int i = 0; i < 8; ++i) op[tid + i * 256] = (bf16)(v[i] * inv);
}

// ---------------------------------------------------------------------------
extern "C" void kernel_launch(void* const* d_in, const int* in_sizes, int n_in,
                              void* d_out, int out_size, void* d_ws, size_t ws_size,
                              hipStream_t stream) {
    constexpr int B = 4, S = 2048, D = 1024;
    constexpr int M = B * S;

    const float* q  = (const float*)d_in[0];
    const float* k  = (const float*)d_in[1];
    const float* v  = (const float*)d_in[2];
    const float* Wq = (const float*)d_in[3];
    const float* bq = (const float*)d_in[4];
    const float* Wk = (const float*)d_in[5];
    const float* bk = (const float*)d_in[6];
    const float* Wv = (const float*)d_in[7];
    const float* bv = (const float*)d_in[8];
    float* out = (float*)d_out;

    // workspace layout
    char*     ws     = (char*)d_ws;
    unsigned* scales = (unsigned*)ws;                       // 3 uints
    bf16*     wt     = (bf16*)(ws + 256);                   // 3 * D*D bf16 (transposed, quantized)
    bf16*     qp     = wt + (size_t)3 * D * D;              // M*D bf16
    bf16*     kp     = qp + (size_t)M * D;                  // M*D bf16
    bf16*     vpT    = kp + (size_t)M * D;                  // M*D bf16, [b][d][s]
    float*    logits = (float*)(vpT + (size_t)M * D);       // S*S f32 (per-batch reuse)
    bf16*     attn   = (bf16*)(logits + (size_t)S * S);     // S*S bf16 (per-batch reuse)

    const float* Wsrc[3] = {Wq, Wk, Wv};

    init_scales_kernel<<<1, 256, 0, stream>>>(scales);
    for (int i = 0; i < 3; ++i)
        absmax_kernel<<<1024, 256, 0, stream>>>(Wsrc[i], scales + i, D * D);
    for (int i = 0; i < 3; ++i)
        quantize_t_kernel<<<(D * D) / 256, 256, 0, stream>>>(
            Wsrc[i], scales + i, wt + (size_t)i * D * D, D);

    // projections: qp/kp row-major bf16, vp stored transposed for AV GEMM
    dim3 gp(D / BN, M / BM);
    gemm_xw_kernel<<<gp, 256, 0, stream>>>(q, wt + (size_t)0 * D * D, bq, qp,  M, D, D, 0);
    gemm_xw_kernel<<<gp, 256, 0, stream>>>(k, wt + (size_t)1 * D * D, bk, kp,  M, D, D, 0);
    gemm_xw_kernel<<<gp, 256, 0, stream>>>(v, wt + (size_t)2 * D * D, bv, vpT, M, D, D, 1);

    const float alpha = 1.0f / 32.0f;  // 1/sqrt(1024)
    for (int b = 0; b < B; ++b) {
        const bf16* qpb  = qp  + (size_t)b * S * D;
        const bf16* kpb  = kp  + (size_t)b * S * D;
        const bf16* vptb = vpT + (size_t)b * D * S;
        float*      outb = out + (size_t)b * S * D;

        gemm_bf16_kernel<<<dim3(S / BN, S / BM), 256, 0, stream>>>(
            qpb, kpb, logits, S, S, D, alpha);
        softmax_kernel<<<S, 256, 0, stream>>>(logits, attn, S);
        gemm_bf16_kernel<<<dim3(D / BN, S / BM), 256, 0, stream>>>(
            attn, vptb, outb, S, D, S, 1.0f);
    }
}
